// MultiheadAttentionwithRelativePositionalEmbedding_71098888618649
// MI455X (gfx1250) — matrix operand order repaired and checked by reading, not compiled, not run on hardware
//
#include <hip/hip_runtime.h>

// ---------------------------------------------------------------------------
// MI455X (gfx1250) multi-head attention w/ relative positional embedding.
// All matmuls on WMMA f32<=f16 matrix cores (wave32). The problem is
// HBM-bound on the attn output tensor (~1 GB traffic @ 23.3 TB/s), so all
// GEMMs use f16 inputs / f32 accumulation, and every bulk store is staged
// through LDS into fully-coalesced b128 stores. Invalid relative-position
// columns are handled with a zero guard row (pointer select, not data masks).
// ---------------------------------------------------------------------------

#define BS      2
#define SEQ     2048
#define DMODEL  512
#define NH      8
#define HD      64
#define ERL     2047      // rows of Er per head
#define MAXLEN  1024

typedef __attribute__((ext_vector_type(16))) _Float16 v16h;
typedef __attribute__((ext_vector_type(8)))  _Float16 v8h;
typedef __attribute__((ext_vector_type(8)))  float    v8f;
typedef __attribute__((ext_vector_type(4)))  float    v4f;

static __device__ __forceinline__ v16h cat8(v8h a, v8h b) {
  v16h r;
#pragma unroll
  for (int i = 0; i < 8; ++i) { r[i] = a[i]; r[i + 8] = b[i]; }
  return r;
}

// A-fragment (16x32 f16), contiguous-K layout matching B:
// lane l holds row m=l%16; lanes 0-15: K=kb+[0..15]; lanes 16-31: K=kb+[16..31].
static __device__ __forceinline__ v16h afrag_f16(const _Float16* row, int kb, int lane) {
  int k0 = kb + ((lane < 16) ? 0 : 16);
  return *(const v16h*)(row + k0);
}

// Same A-fragment but converting from f32 source on the fly (v_cvt_pk_f16_f32).
static __device__ __forceinline__ v16h afrag_f32(const float* row, int kb, int lane) {
  int k0 = kb + ((lane < 16) ? 0 : 16);
  v16h r;
#pragma unroll
  for (int i = 0; i < 8; ++i) {
    r[i]     = (_Float16)row[k0 + i];
    r[i + 8] = (_Float16)row[k0 + 8 + i];
  }
  return r;
}

// B-fragment (32x16 f16): lane l holds column n=l%16, K=kb+[0..15] (lanes 0-15)
// or kb+[16..31] (lanes 16-31). `col` points at the K-contiguous column storage.
static __device__ __forceinline__ v16h bfrag_f16(const _Float16* col, int kb, int lane) {
  int k0 = kb + ((lane < 16) ? 0 : 16);
  return *(const v16h*)(col + k0);
}

static __device__ __forceinline__ v8f wmma(v16h a, v16h b, v8f c) {
  return __builtin_amdgcn_wmma_f32_16x16x32_f16(false, a, false, b, (short)0, c,
                                                false, false);
}

// ---------------------------------------------------------------------------
// Kernel 1: Er f32 -> f16, plus one zeroed 64-half guard row at the end
// (used as the target for out-of-band relative positions).
// ---------------------------------------------------------------------------
__global__ void er_convert_kernel(const float* __restrict__ er,
                                  _Float16* __restrict__ erf) {
  const int N = NH * ERL * HD;
  int i = blockIdx.x * blockDim.x + threadIdx.x;
  if (i < N)            erf[i] = (_Float16)er[i];
  else if (i < N + HD)  erf[i] = (_Float16)0.0f;     // zero guard row
}

// ---------------------------------------------------------------------------
// Kernel 2: projection Y = X@W + b, templated on which matrix so control flow
// is uniform. WHICH: 0 -> q [b,h,t,hd], 1 -> k [b,h,t,hd], 2 -> v transposed
// [b,h,hd,s]. Epilogue staged through LDS -> coalesced b128 stores.
// ---------------------------------------------------------------------------
template <int WHICH>
__global__ __launch_bounds__(128)
void proj_kernel(const float* __restrict__ X, const float* __restrict__ W,
                 const float* __restrict__ bias, _Float16* __restrict__ OUT) {
  __shared__ _Float16 Wt[64][32];        // transposed W tile: Wt[n][k]  (4 KB)
  __shared__ _Float16 stg[4][16 * 64];   // per-wave output tile        (8 KB)

  const int tid = threadIdx.x, lane = tid & 31, wave = tid >> 5;
  const int m0 = blockIdx.y * 64 + wave * 16;   // wave-private 16 output rows
  const int n0 = blockIdx.x * 64;

  v8f acc[4] = {};
  const float* xrow = X + (size_t)(m0 + (lane & 15)) * DMODEL;

  for (int kb = 0; kb < DMODEL; kb += 32) {
    __syncthreads();
    {   // stage W[kb..kb+32)[n0..n0+64) transposed into LDS (16 elems/thread)
      int e = tid * 16;
      int k = e >> 6;          // 0..31
      int n = e & 63;          // 0,16,32,48
      const float* wr = W + (size_t)(kb + k) * DMODEL + n0 + n;
#pragma unroll
      for (int i = 0; i < 16; ++i) Wt[n + i][k] = (_Float16)wr[i];
    }
    __syncthreads();

    __builtin_prefetch(xrow + kb + 64, 0, 0);      // global_prefetch_b8
    v16h a = afrag_f32(xrow, kb, lane);
#pragma unroll
    for (int s = 0; s < 4; ++s) {
      v16h b = bfrag_f16(&Wt[s * 16 + (lane & 15)][0], 0, lane);
      acc[s] = wmma(a, b, acc[s]);
    }
  }

  // ---- epilogue: bias + f16 convert into per-wave LDS tile ----
  _Float16* sw = stg[wave];
#pragma unroll
  for (int s = 0; s < 4; ++s) {
    int nl = s * 16 + (lane & 15);
    float bval = bias[n0 + nl];
#pragma unroll
    for (int r = 0; r < 8; ++r) {
      int ml = r + ((lane < 16) ? 0 : 8);     // C/D layout: lanes>=16 -> M+8
      _Float16 hv = (_Float16)(acc[s][r] + bval);
      if (WHICH == 2) sw[nl * 16 + ml] = hv;  // transposed for v
      else            sw[ml * 64 + nl] = hv;
    }
  }
  // same-wave LDS ops are in-order -> no barrier needed before re-reading.

  const int bb = m0 >> 11;            // row / SEQ
  const int t0 = m0 & (SEQ - 1);
  const int h  = n0 >> 6;             // 64-wide n-tile == one head
  if (WHICH == 2) {
    // v transposed: 64 hd-rows x 16 t-cols; each row contiguous 16 halves.
    _Float16* dstb = OUT + (size_t)((bb * NH + h) * HD) * SEQ + t0;
#pragma unroll
    for (int i = 0; i < 2; ++i) {
      int hd = lane + i * 32;
      const _Float16* srcr = sw + hd * 16;
      _Float16* d = dstb + (size_t)hd * SEQ;
      *(v8h*)(d)     = *(const v8h*)(srcr);
      *(v8h*)(d + 8) = *(const v8h*)(srcr + 8);
    }
  } else {
    // q/k: the 16x64 tile is one contiguous 1024-half region (row stride==64).
    _Float16* dst = OUT + ((size_t)((bb * NH + h) * SEQ) + t0) * HD;
#pragma unroll
    for (int i = 0; i < 4; ++i) {
      int off = (lane + i * 32) * 8;          // 128 chunks of 8 halves
      *(v8h*)(dst + off) = *(const v8h*)(sw + off);
    }
  }
}

// ---------------------------------------------------------------------------
// Kernel 3: raw scores = (q @ k^T + Srel) / sqrt(HD), written to attn buffer.
// Srel fused via per-tile 16x32 diagonal GEMM G = q_tile @ Er_t^T and a
// same-wave LDS gather G[i][j-i+15]. Consecutive 16x16 sub-tiles share 16 of
// their 32 diagonal columns, so the high fragment pair is carried over.
// Invalid columns load from the zero guard row (pointer select only).
// ---------------------------------------------------------------------------
__global__ __launch_bounds__(128)
void scores_kernel(const _Float16* __restrict__ qf, const _Float16* __restrict__ kf,
                   const _Float16* __restrict__ erf, const int* __restrict__ layer_p,
                   float* __restrict__ attn) {
  __shared__ float gbuf[4][16 * 32];   //  8 KB: diagonal-GEMM scratch
  __shared__ float sbuf[4][16 * 64];   // 16 KB: per-wave score tile

  const int tid = threadIdx.x, lane = tid & 31, wave = tid >> 5;
  const int bh = blockIdx.z;
  const int h  = bh & (NH - 1);
  const int t0 = blockIdx.y * 64 + wave * 16;
  const int s0 = blockIdx.x * 64;

  const int layer = *layer_p;           // static scalar (uniform s_load)
  const int step  = 1 << layer;
  int dl = 1 + ((SEQ - 1) >> layer);
  if (dl > MAXLEN) dl = MAXLEN;

  const _Float16* qrow = qf + ((size_t)(bh * SEQ) + t0 + (lane & 15)) * HD;
  v16h a0 = afrag_f16(qrow, 0,  lane);  // reused by both q@k^T and q@Er^T
  v16h a1 = afrag_f16(qrow, 32, lane);

  const _Float16* zrow = erf + (size_t)NH * ERL * HD;   // zero guard row
  const _Float16* ehead = erf + (size_t)h * ERL * HD;

  // Load the 16-column diagonal fragment group whose first column is gb
  // (gb in Er_t column space, i.e. l = gb + lane%16).
  auto load_group = [&](int gb, v16h& bK0, v16h& bK32) {
    int rel = gb + (lane & 15) - 2047;                 // s - t
    bool aligned = ((rel & (step - 1)) == 0);
    int  rr = rel >> layer;
    bool valid = aligned && (rr > -dl) && (rr < dl);
    const _Float16* ecol =
        valid ? (ehead + (size_t)((MAXLEN - 1) + rr) * HD) : zrow;
    bK0  = bfrag_f16(ecol, 0,  lane);
    bK32 = bfrag_f16(ecol, 32, lane);
  };

  float* gl = gbuf[wave];
  float* sb = sbuf[wave];

  const int lbase0 = s0 - t0 + 2047 - 15;   // first diagonal column of sub 0
  v16h eA0, eA32;                           // low 16-column group (carried)
  load_group(lbase0, eA0, eA32);

#pragma unroll
  for (int sub = 0; sub < 4; ++sub) {
    const int sg = s0 + sub * 16;

    // ---- q @ k^T (K = 64 -> 2 WMMA) ----
    const _Float16* kcol = kf + ((size_t)(bh * SEQ) + sg + (lane & 15)) * HD;
    v8f sacc = {};
    sacc = wmma(a0, bfrag_f16(kcol, 0,  lane), sacc);
    sacc = wmma(a1, bfrag_f16(kcol, 32, lane), sacc);

    // ---- G = q_tile @ Er_t^T over columns [lbase, lbase+32) ----
    v16h eB0, eB32;                       // high group; becomes next sub's low
    load_group(lbase0 + sub * 16 + 16, eB0, eB32);

    v8f g0 = {}, g1 = {};
    g0 = wmma(a0, eA0,  g0);  g0 = wmma(a1, eA32, g0);
    g1 = wmma(a0, eB0,  g1);  g1 = wmma(a1, eB32, g1);
    eA0 = eB0; eA32 = eB32;

    // stage G -> LDS (same-wave, in-order), then diagonal gather into sbuf
#pragma unroll
    for (int r = 0; r < 8; ++r) {
      int i = r + ((lane < 16) ? 0 : 8);
      int c = lane & 15;
      gl[i * 32 + c]      = g0[r];
      gl[i * 32 + 16 + c] = g1[r];
    }
#pragma unroll
    for (int r = 0; r < 8; ++r) {
      int i = r + ((lane < 16) ? 0 : 8);
      int j = lane & 15;
      float srel = gl[i * 32 + (j - i + 15)];
      sb[i * 64 + sub * 16 + j] = (sacc[r] + srel) * 0.125f;  // 1/sqrt(64)
    }
  }

  // ---- coalesced bulk store: 16 rows x 64 floats, rows stride SEQ ----
  float* dst = attn + ((size_t)(bh * SEQ) + t0) * SEQ + s0;
#pragma unroll
  for (int i = 0; i < 8; ++i) {
    int chunk = lane + i * 32;        // 256 chunks of 4 floats
    int rr = chunk >> 4;              // 16 chunks per row
    int cc = (chunk & 15) * 4;
    *(v4f*)(dst + (size_t)rr * SEQ + cc) = *(const v4f*)(sb + rr * 64 + cc);
  }
}

// ---------------------------------------------------------------------------
// Kernel 4: row-wise softmax over SEQ (in place in the attn output buffer).
// ---------------------------------------------------------------------------
__global__ __launch_bounds__(256)
void softmax_kernel(float* __restrict__ attn) {
  __shared__ float red[256];
  float* p = attn + (size_t)blockIdx.x * SEQ;
  const int tid = threadIdx.x;

  float vals[8];
  float m = -1e30f;
#pragma unroll
  for (int i = 0; i < 8; ++i) { vals[i] = p[tid + i * 256]; m = fmaxf(m, vals[i]); }
  red[tid] = m; __syncthreads();
  for (int off = 128; off > 0; off >>= 1) {
    if (tid < off) red[tid] = fmaxf(red[tid], red[tid + off]);
    __syncthreads();
  }
  m = red[0]; __syncthreads();

  float s = 0.f;
#pragma unroll
  for (int i = 0; i < 8; ++i) { vals[i] = __expf(vals[i] - m); s += vals[i]; }
  red[tid] = s; __syncthreads();
  for (int off = 128; off > 0; off >>= 1) {
    if (tid < off) red[tid] += red[tid + off];
    __syncthreads();
  }
  float inv = 1.0f / red[0];
#pragma unroll
  for (int i = 0; i < 8; ++i) p[tid + i * 256] = vals[i] * inv;
}

// ---------------------------------------------------------------------------
// Kernel 5: out = attn @ v  (attn f32 -> f16 fragments; v pre-transposed).
// Output tile staged in LDS for coalesced b128 stores.
// ---------------------------------------------------------------------------
__global__ __launch_bounds__(128)
void av_kernel(const float* __restrict__ attn, const _Float16* __restrict__ vtf,
               float* __restrict__ out) {
  __shared__ float obuf[4][16 * 64];   // 16 KB

  const int tid = threadIdx.x, lane = tid & 31, wave = tid >> 5;
  const int bh = blockIdx.y;
  const int b = bh >> 3, h = bh & 7;
  const int t0 = blockIdx.x * 64 + wave * 16;

  const float* arow = attn + ((size_t)(bh * SEQ) + t0 + (lane & 15)) * SEQ;
  v8f acc[4] = {};
  for (int kb = 0; kb < SEQ; kb += 32) {
    __builtin_prefetch(arow + kb + 64, 0, 0);
    v16h a = afrag_f32(arow, kb, lane);
#pragma unroll
    for (int s = 0; s < 4; ++s) {
      const _Float16* vcol = vtf + ((size_t)(bh * HD) + s * 16 + (lane & 15)) * SEQ;
      acc[s] = wmma(a, bfrag_f16(vcol, kb, lane), acc[s]);
    }
  }

  float* ob = obuf[wave];
#pragma unroll
  for (int s = 0; s < 4; ++s) {
#pragma unroll
    for (int r = 0; r < 8; ++r) {
      int ml = r + ((lane < 16) ? 0 : 8);
      ob[ml * 64 + s * 16 + (lane & 15)] = acc[s][r];
    }
  }
  // coalesced store: 16 rows x 64 floats, rows stride DMODEL, col base h*HD
  float* dst = out + ((size_t)(b * SEQ) + t0) * DMODEL + h * HD;
#pragma unroll
  for (int i = 0; i < 8; ++i) {
    int chunk = lane + i * 32;
    int rr = chunk >> 4;
    int cc = (chunk & 15) * 4;
    *(v4f*)(dst + (size_t)rr * DMODEL + cc) = *(const v4f*)(ob + rr * 64 + cc);
  }
}

// ---------------------------------------------------------------------------
extern "C" void kernel_launch(void* const* d_in, const int* in_sizes, int n_in,
                              void* d_out, int out_size, void* d_ws, size_t ws_size,
                              hipStream_t stream) {
  const float* query = (const float*)d_in[0];
  const float* key_  = (const float*)d_in[1];
  const float* value = (const float*)d_in[2];
  const float* Wq    = (const float*)d_in[3];
  const float* bq    = (const float*)d_in[4];
  const float* Wk    = (const float*)d_in[5];
  const float* bk    = (const float*)d_in[6];
  const float* Wv    = (const float*)d_in[7];
  const float* bv    = (const float*)d_in[8];
  const float* Er    = (const float*)d_in[9];
  const int*   layer = (const int*)d_in[10];

  float* out  = (float*)d_out;
  float* attn = out + (size_t)BS * SEQ * DMODEL;     // outputs concatenated

  const size_t QN = (size_t)BS * NH * SEQ * HD;      // 2,097,152 halves each
  _Float16* qf  = (_Float16*)d_ws;
  _Float16* kf  = qf + QN;
  _Float16* vtf = kf + QN;
  _Float16* erf = vtf + QN;        // NH*ERL*HD halves + HD-half zero guard row

  er_convert_kernel<<<dim3((NH * ERL * HD + HD + 255) / 256), 256, 0, stream>>>(
      Er, erf);

  dim3 pgrid(DMODEL / 64, (BS * SEQ) / 64);
  proj_kernel<0><<<pgrid, 128, 0, stream>>>(query, Wq, bq, qf);
  proj_kernel<1><<<pgrid, 128, 0, stream>>>(key_,  Wk, bk, kf);
  proj_kernel<2><<<pgrid, 128, 0, stream>>>(value, Wv, bv, vtf);

  scores_kernel<<<dim3(SEQ / 64, SEQ / 64, BS * NH), 128, 0, stream>>>(
      qf, kf, erf, layer, attn);

  softmax_kernel<<<dim3(BS * NH * SEQ), 256, 0, stream>>>(attn);

  av_kernel<<<dim3(SEQ / 64, BS * NH), 128, 0, stream>>>(attn, vtf, out);
}
